// NetlistGNN_86921548136520
// MI455X (gfx1250) — matrix operand order repaired and checked by reading, not compiled, not run on hardware
//
#include <hip/hip_runtime.h>

#define N_NODES 50000
#define N_NETS  10000
#define E_PIN   100000
#define E_NEAR  150000
#define KDIM    544    // 512 bilinear (32x16 outer product) + 32 bias-fold rows
#define KTILES  17     // KDIM / 32
#define WPELEMS (KTILES * 2 * 32 * 16)   // fragment-ordered Wp: 17408 bf16

typedef __attribute__((ext_vector_type(16))) __bf16 v16bf;
typedef __attribute__((ext_vector_type(8)))  float  v8f;

__device__ __forceinline__ unsigned short f2bf(float f) {
    union { float f; unsigned u; } v; v.f = f;
    unsigned u = v.u;
    u += 0x7FFFu + ((u >> 16) & 1u);     // round-to-nearest-even
    return (unsigned short)(u >> 16);
}

// ---------------------------------------------------------------------------
// Fused NNConv message kernel: per edge e,
//   msg[e,:] = x_e (32) contracted with We_e (32x32), We_e = p_e(16) @ W + b
// expressed as GEMM  z(E x 544) @ Wp(544 x 32), z built on the fly in regs.
// One wave per 16-edge tile; 2x v_wmma_f32_16x16x32_bf16 per K step.
// A-fragment structure: for K-tile kt, element j   = x[2kt]  *p[half*8+j],
//                                      element 8+j = x[2kt+1]*p[half*8+j].
// ---------------------------------------------------------------------------
__global__ __launch_bounds__(256)
void nnconv_wmma_kernel(const float* __restrict__ xfeat,   // [nx, 32] gathered via xidx
                        const float* __restrict__ pfeat,   // [E, 16]
                        const int*   __restrict__ xidx,    // [E] gather index into xfeat
                        const int*   __restrict__ sidx,    // [E] scatter index (dst node)
                        const unsigned short* __restrict__ Wp, // fragment-ordered [17][2][32][16]
                        float* __restrict__ accum,         // [N_NODES*32] pre-zeroed
                        int nEdges)
{
    __shared__ __align__(16) unsigned short sWp[WPELEMS]; // 34816 B, fragment-ordered
    __shared__ float sX[8][16][32];                       // 16384 B
    __shared__ float sP[8][16][16];                       //  8192 B

    const int tid  = threadIdx.x;
    const int wave = tid >> 5;
    const int lane = tid & 31;
    const int half = lane >> 4;   // 0: lanes 0-15, 1: lanes 16-31
    const int l16  = lane & 15;

    // Stage fragment-ordered weights block-wide (vector copy: 17408 u16 = 8704 dwords)
    {
        const unsigned* src = (const unsigned*)Wp;
        unsigned* dst = (unsigned*)sWp;
        for (int i = tid; i < WPELEMS / 2; i += 256) dst[i] = src[i];
    }

    const int nTiles = nEdges >> 4;
    const int tile   = blockIdx.x * 8 + wave;
    const int e0     = tile << 4;

    if (tile < nTiles) {
        for (int i = lane; i < 16 * 32; i += 32) {
            int er = i >> 5, c = i & 31;
            sX[wave][er][c] = xfeat[(size_t)xidx[e0 + er] * 32 + c];
        }
        for (int i = lane; i < 16 * 16; i += 32) {
            int er = i >> 4, c = i & 15;
            sP[wave][er][c] = pfeat[(size_t)(e0 + er) * 16 + c];
        }
    }
    __syncthreads();
    if (tile >= nTiles) return;

    // Hoist this lane's A-matrix row (edge l16) into registers.
    float xr[32];
    #pragma unroll
    for (int i = 0; i < 8; ++i) {
        float4 q = ((const float4*)sX[wave][l16])[i];
        xr[4 * i + 0] = q.x; xr[4 * i + 1] = q.y;
        xr[4 * i + 2] = q.z; xr[4 * i + 3] = q.w;
    }
    float pr[8];
    #pragma unroll
    for (int j = 0; j < 8; ++j) pr[j] = sP[wave][l16][half * 8 + j];

    v8f c0 = {}; v8f c1 = {};

    // Main bilinear K loop (kt = 0..15)
    for (int kt = 0; kt < 16; ++kt) {
        union { v16bf v; unsigned short u[16]; } A;
        const float x1 = xr[2 * kt];
        const float x2 = xr[2 * kt + 1];
        #pragma unroll
        for (int j = 0; j < 8; ++j) {
            A.u[j]     = f2bf(x1 * pr[j]);
            A.u[8 + j] = f2bf(x2 * pr[j]);
        }
        union { v16bf v; uint4 q[2]; } B0, B1;
        const uint4* bp = (const uint4*)&sWp[((kt * 2) * 32 + lane) * 16];
        B0.q[0] = bp[0];  B0.q[1] = bp[1];     // ntile 0: 32B contiguous per lane
        B1.q[0] = bp[64]; B1.q[1] = bp[65];    // ntile 1: +32 lanes * 16 u16 = +64 uint4
        c0 = __builtin_amdgcn_wmma_f32_16x16x32_bf16(false, A.v, false, B0.v,
                                                     (short)0, c0, false, false);
        c1 = __builtin_amdgcn_wmma_f32_16x16x32_bf16(false, A.v, false, B1.v,
                                                     (short)0, c1, false, false);
    }

    // Bias-fold tile (kt = 16): A element j = bf16(x[half*8+j]), 8+j = bf16(x[16+half*8+j])
    {
        union { v16bf v; unsigned short u[16]; } A;
        #pragma unroll
        for (int j = 0; j < 8; ++j) {
            A.u[j]     = f2bf(xr[half * 8 + j]);
            A.u[8 + j] = f2bf(xr[16 + half * 8 + j]);
        }
        union { v16bf v; uint4 q[2]; } B0, B1;
        const uint4* bp = (const uint4*)&sWp[((16 * 2) * 32 + lane) * 16];
        B0.q[0] = bp[0];  B0.q[1] = bp[1];
        B1.q[0] = bp[64]; B1.q[1] = bp[65];
        c0 = __builtin_amdgcn_wmma_f32_16x16x32_bf16(false, A.v, false, B0.v,
                                                     (short)0, c0, false, false);
        c1 = __builtin_amdgcn_wmma_f32_16x16x32_bf16(false, A.v, false, B1.v,
                                                     (short)0, c1, false, false);
    }

    // D layout (f32 16x16): VGPR r -> row r + 8*half, col l16
    #pragma unroll
    for (int r = 0; r < 8; ++r) {
        int row = r + half * 8;
        int dst = sidx[e0 + row];
        atomicAdd(&accum[(size_t)dst * 32 + l16],      c0[r]);
        atomicAdd(&accum[(size_t)dst * 32 + 16 + l16], c1[r]);
    }
}

// Pre-pack W' into fragment-ordered bf16: [kt][ntile][lane][j]
//   kk  = kt*32 + (j<8 ? 8*(lane>>4)+j : 16+8*(lane>>4)+(j-8))     (K row in the 544-deep GEMM)
//   col = ntile*16 + (lane&15)             (output column)
// rows 0-511: bilinear weights W[k,i*32+col] with kk = i*16+k; rows 512-543: bias fold
__global__ void prep_wp_kernel(const float* __restrict__ w,  // [16,1024]
                               const float* __restrict__ b,  // [1024]
                               unsigned short* __restrict__ Wp)
{
    int t = blockIdx.x * blockDim.x + threadIdx.x;
    if (t >= WPELEMS) return;
    int j     = t & 15;
    int lane  = (t >> 4) & 31;
    int ntile = (t >> 9) & 1;
    int kt    = t >> 10;
    int kk    = kt * 32 + ((j < 8) ? (lane >> 4) * 8 + j : 16 + (lane >> 4) * 8 + (j - 8));
    int col   = ntile * 16 + (lane & 15);
    float val;
    if (kk < 512) { int i = kk >> 4, k = kk & 15; val = w[k * 1024 + i * 32 + col]; }
    else          { val = b[(kk - 512) * 32 + col]; }
    Wp[t] = f2bf(val);
}

// Generic encoder: out = leaky_relu(in @ W + b, 0.01)
__global__ void enc_kernel(const float* __restrict__ in, const float* __restrict__ w,
                           const float* __restrict__ b, float* __restrict__ out,
                           int rows, int din, int dout)
{
    int t = blockIdx.x * blockDim.x + threadIdx.x;
    if (t >= rows * dout) return;
    int r = t / dout, c = t % dout;
    float acc = b[c];
    for (int k = 0; k < din; ++k) acc += in[(size_t)r * din + k] * w[k * dout + c];
    out[t] = acc > 0.0f ? acc : 0.01f * acc;
}

__global__ void deg_kernel(const int* __restrict__ idx, float* __restrict__ deg, int n)
{
    int t = blockIdx.x * blockDim.x + threadIdx.x;
    if (t < n) atomicAdd(&deg[idx[t]], 1.0f);   // exact integer sums -> deterministic
}

__global__ void norm_kernel(const float* __restrict__ deg, float* __restrict__ rsq,
                            float* __restrict__ inv, int n)
{
    int t = blockIdx.x * blockDim.x + threadIdx.x;
    if (t >= n) return;
    float d = fmaxf(deg[t], 1.0f);
    rsq[t] = rsqrtf(d);
    inv[t] = 1.0f / d;
}

// GraphConv scatter: agg[dst] += node_f[src]*norm_s[src]
__global__ void pins_scatter_kernel(const float* __restrict__ node_f,
                                    const float* __restrict__ norm_s,
                                    const int* __restrict__ src, const int* __restrict__ dst,
                                    float* __restrict__ agg, int E)
{
    int t = blockIdx.x * blockDim.x + threadIdx.x;
    if (t >= E * 32) return;
    int e = t >> 5, c = t & 31;
    int s = src[e];
    atomicAdd(&agg[(size_t)dst[e] * 32 + c], node_f[(size_t)s * 32 + c] * norm_s[s]);
}

// net_new = (agg * norm_d) @ pins_w + pins_b   (no activation)
__global__ void net_update_kernel(const float* __restrict__ agg, const float* __restrict__ norm_d,
                                  const float* __restrict__ w, const float* __restrict__ b,
                                  float* __restrict__ out, int n)
{
    int t = blockIdx.x * blockDim.x + threadIdx.x;
    if (t >= n * 32) return;
    int r = t >> 5, c = t & 31;
    float nd = norm_d[r];
    float acc = b[c];
    for (int k = 0; k < 32; ++k) acc += agg[(size_t)r * 32 + k] * nd * w[k * 32 + c];
    out[t] = acc;
}

// node_f = max(pinned*invP + pb, near*invN + nb)
__global__ void combine_kernel(const float* __restrict__ pA, const float* __restrict__ nA,
                               const float* __restrict__ invP, const float* __restrict__ invN,
                               const float* __restrict__ pb, const float* __restrict__ nb,
                               float* __restrict__ out, int n)
{
    int t = blockIdx.x * blockDim.x + threadIdx.x;
    if (t >= n * 32) return;
    int r = t >> 5, c = t & 31;
    float p = pA[t] * invP[r] + pb[c];
    float q = nA[t] * invN[r] + nb[c];
    out[t] = fmaxf(p, q);
}

// h1 = tanh(concat(in_node[8], node_f[32]) @ W(40x32) + b)
__global__ void head1_kernel(const float* __restrict__ in_node, const float* __restrict__ node_f,
                             const float* __restrict__ w, const float* __restrict__ b,
                             float* __restrict__ out, int n)
{
    int t = blockIdx.x * blockDim.x + threadIdx.x;
    if (t >= n * 32) return;
    int r = t >> 5, c = t & 31;
    float acc = b[c];
    for (int k = 0; k < 8; ++k)  acc += in_node[(size_t)r * 8 + k]  * w[k * 32 + c];
    for (int k = 0; k < 32; ++k) acc += node_f[(size_t)r * 32 + k]  * w[(8 + k) * 32 + c];
    out[t] = tanhf(acc);
}

__global__ void head2_kernel(const float* __restrict__ h, const float* __restrict__ w,
                             const float* __restrict__ b, float* __restrict__ out, int n)
{
    int t = blockIdx.x * blockDim.x + threadIdx.x;
    if (t >= n * 32) return;
    int r = t >> 5, c = t & 31;
    float acc = b[c];
    for (int k = 0; k < 32; ++k) acc += h[(size_t)r * 32 + k] * w[k * 32 + c];
    out[t] = tanhf(acc);
}

__global__ void head3_kernel(const float* __restrict__ h, const float* __restrict__ w,
                             const float* __restrict__ b, float* __restrict__ out, int n)
{
    int t = blockIdx.x * blockDim.x + threadIdx.x;
    if (t >= n * 2) return;
    int r = t >> 1, c = t & 1;
    float acc = b[c];
    for (int k = 0; k < 32; ++k) acc += h[(size_t)r * 32 + k] * w[k * 2 + c];
    out[t] = 1.0f / (1.0f + expf(-acc));
}

// ---------------------------------------------------------------------------
extern "C" void kernel_launch(void* const* d_in, const int* in_sizes, int n_in,
                              void* d_out, int out_size, void* d_ws, size_t ws_size,
                              hipStream_t stream)
{
    (void)in_sizes; (void)n_in; (void)out_size; (void)ws_size;

    const float* in_node  = (const float*)d_in[0];
    const float* in_net   = (const float*)d_in[1];
    const float* in_pin   = (const float*)d_in[2];
    const float* in_edge  = (const float*)d_in[3];
    const int*   pins_src = (const int*)d_in[4];
    const int*   pins_dst = (const int*)d_in[5];
    const int*   near_src = (const int*)d_in[6];
    const int*   near_dst = (const int*)d_in[7];

    // params: JAX pytree order (dict keys sorted; tuples/lists in order)
    const float* edge_w = (const float*)d_in[8];
    const float* edge_b = (const float*)d_in[9];
    const float* L_geom_b[2]   = {(const float*)d_in[10], (const float*)d_in[18]};
    const float* L_geom_w[2]   = {(const float*)d_in[11], (const float*)d_in[19]};
    const float* L_near_b[2]   = {(const float*)d_in[12], (const float*)d_in[20]};
    const float* L_pinned_b[2] = {(const float*)d_in[13], (const float*)d_in[21]};
    const float* L_pins_b[2]   = {(const float*)d_in[14], (const float*)d_in[22]};
    const float* L_pins_w[2]   = {(const float*)d_in[15], (const float*)d_in[23]};
    const float* L_topo_b[2]   = {(const float*)d_in[16], (const float*)d_in[24]};
    const float* L_topo_w[2]   = {(const float*)d_in[17], (const float*)d_in[25]};
    const float* net_w  = (const float*)d_in[26];
    const float* net_b  = (const float*)d_in[27];
    const float* node_w = (const float*)d_in[28];
    const float* node_b = (const float*)d_in[29];
    const float* out1_b = (const float*)d_in[30];
    const float* out1_w = (const float*)d_in[31];
    const float* out2_b = (const float*)d_in[32];
    const float* out2_w = (const float*)d_in[33];
    const float* out3_b = (const float*)d_in[34];
    const float* out3_w = (const float*)d_in[35];
    const float* pin_w  = (const float*)d_in[36];
    const float* pin_b  = (const float*)d_in[37];

    // ---- workspace carve ----
    float* ws = (float*)d_ws;
    size_t off = 0;
    auto alloc = [&](size_t nelem) { float* p = ws + off; off += nelem; return p; };
    float* node_fa   = alloc((size_t)N_NODES * 32);
    float* node_fb   = alloc((size_t)N_NODES * 32);
    float* net_fa    = alloc((size_t)N_NETS * 32);
    float* net_fb    = alloc((size_t)N_NETS * 32);
    float* pin_f     = alloc((size_t)E_PIN * 16);
    float* edge_f    = alloc((size_t)E_NEAR * 16);
    float* deg_no    = alloc(N_NODES);
    float* deg_ni    = alloc(N_NETS);
    float* deg_nr    = alloc(N_NODES);
    float* norm_s    = alloc(N_NODES);
    float* inv_pin   = alloc(N_NODES);
    float* norm_d    = alloc(N_NETS);
    float* inv_dmy   = alloc(N_NETS);
    float* rsq_dmy   = alloc(N_NODES);
    float* inv_near  = alloc(N_NODES);
    float* agg       = alloc((size_t)N_NETS * 32);
    float* pinnedA   = alloc((size_t)N_NODES * 32);
    float* nearA     = alloc((size_t)N_NODES * 32);
    unsigned short* wpBase = (unsigned short*)(ws + off);
    unsigned short* WpT[2]; unsigned short* WpG[2];
    for (int l = 0; l < 2; ++l) {
        WpT[l] = wpBase; wpBase += WPELEMS;
        WpG[l] = wpBase; wpBase += WPELEMS;
    }

    #define NB(n) (((n) + 255) / 256)

    // ---- feature encoders ----
    enc_kernel<<<NB(N_NODES * 32), 256, 0, stream>>>(in_node, node_w, node_b, node_fa, N_NODES, 8, 32);
    enc_kernel<<<NB(N_NETS * 32),  256, 0, stream>>>(in_net,  net_w,  net_b,  net_fa,  N_NETS,  8, 32);
    enc_kernel<<<NB(E_PIN * 16),   256, 0, stream>>>(in_pin,  pin_w,  pin_b,  pin_f,   E_PIN,   8, 16);
    enc_kernel<<<NB(E_NEAR * 16),  256, 0, stream>>>(in_edge, edge_w, edge_b, edge_f,  E_NEAR,  4, 16);

    // ---- degrees / norms ----
    hipMemsetAsync(deg_no, 0, (size_t)N_NODES * 4, stream);
    hipMemsetAsync(deg_ni, 0, (size_t)N_NETS * 4, stream);
    hipMemsetAsync(deg_nr, 0, (size_t)N_NODES * 4, stream);
    deg_kernel<<<NB(E_PIN),  256, 0, stream>>>(pins_src, deg_no, E_PIN);
    deg_kernel<<<NB(E_PIN),  256, 0, stream>>>(pins_dst, deg_ni, E_PIN);
    deg_kernel<<<NB(E_NEAR), 256, 0, stream>>>(near_dst, deg_nr, E_NEAR);
    norm_kernel<<<NB(N_NODES), 256, 0, stream>>>(deg_no, norm_s, inv_pin, N_NODES);
    norm_kernel<<<NB(N_NETS),  256, 0, stream>>>(deg_ni, norm_d, inv_dmy, N_NETS);
    norm_kernel<<<NB(N_NODES), 256, 0, stream>>>(deg_nr, rsq_dmy, inv_near, N_NODES);

    // ---- pre-pack NNConv weights to fragment-ordered bf16 (bias folded into K) ----
    for (int l = 0; l < 2; ++l) {
        prep_wp_kernel<<<NB(WPELEMS), 256, 0, stream>>>(L_topo_w[l], L_topo_b[l], WpT[l]);
        prep_wp_kernel<<<NB(WPELEMS), 256, 0, stream>>>(L_geom_w[l], L_geom_b[l], WpG[l]);
    }

    // ---- message-passing layers ----
    float* node_cur = node_fa; float* node_nxt = node_fb;
    float* net_cur  = net_fa;  float* net_nxt  = net_fb;
    for (int l = 0; l < 2; ++l) {
        hipMemsetAsync(agg,     0, (size_t)N_NETS * 32 * 4, stream);
        hipMemsetAsync(pinnedA, 0, (size_t)N_NODES * 32 * 4, stream);
        hipMemsetAsync(nearA,   0, (size_t)N_NODES * 32 * 4, stream);

        // 'pins': GraphConv node -> net
        pins_scatter_kernel<<<NB(E_PIN * 32), 256, 0, stream>>>(node_cur, norm_s, pins_src, pins_dst, agg, E_PIN);
        net_update_kernel<<<NB(N_NETS * 32), 256, 0, stream>>>(agg, norm_d, L_pins_w[l], L_pins_b[l], net_nxt, N_NETS);

        // 'pinned': NNConv net -> node (WMMA)
        {
            int tiles = E_PIN / 16;
            nnconv_wmma_kernel<<<(tiles + 7) / 8, 256, 0, stream>>>(
                net_cur, pin_f, pins_dst, pins_src, WpT[l], pinnedA, E_PIN);
        }
        // 'near': NNConv node -> node (WMMA)
        {
            int tiles = E_NEAR / 16;
            nnconv_wmma_kernel<<<(tiles + 7) / 8, 256, 0, stream>>>(
                node_cur, edge_f, near_src, near_dst, WpG[l], nearA, E_NEAR);
        }
        combine_kernel<<<NB(N_NODES * 32), 256, 0, stream>>>(
            pinnedA, nearA, inv_pin, inv_near, L_pinned_b[l], L_near_b[l], node_nxt, N_NODES);

        float* t;
        t = node_cur; node_cur = node_nxt; node_nxt = t;
        t = net_cur;  net_cur  = net_nxt;  net_nxt  = t;
    }

    // ---- output MLP (reuse accumulators as hidden buffers) ----
    float* h1 = pinnedA;
    float* h2 = nearA;
    head1_kernel<<<NB(N_NODES * 32), 256, 0, stream>>>(in_node, node_cur, out1_w, out1_b, h1, N_NODES);
    head2_kernel<<<NB(N_NODES * 32), 256, 0, stream>>>(h1, out2_w, out2_b, h2, N_NODES);
    head3_kernel<<<NB(N_NODES * 2),  256, 0, stream>>>(h2, out3_w, out3_b, (float*)d_out, N_NODES);
}
